// LabelScorer_28870770163774
// MI455X (gfx1250) — matrix operand order repaired and checked by reading, not compiled, not run on hardware
//
#include <hip/hip_runtime.h>
#include <hip/hip_bf16.h>

typedef unsigned short u16;
typedef __attribute__((ext_vector_type(16))) __bf16 v16bf;
typedef __attribute__((ext_vector_type(8)))  float  v8f;

union FragBf {
    v16bf v;
    uint4 q[2];
};

__device__ inline u16 f2bf(float f) {
    unsigned int u = __float_as_uint(f);
    unsigned int r = (u + 0x7FFFu + ((u >> 16) & 1u)) >> 16;  // RNE
    return (u16)r;
}

// ---------------------------------------------------------------------------
// fp32 -> bf16 vectorized convert (8 elements / thread)
// ---------------------------------------------------------------------------
__global__ void convv_kernel(const float* __restrict__ src, u16* __restrict__ dst, int n8)
{
    int i = blockIdx.x * blockDim.x + threadIdx.x;
    if (i >= n8) return;
    const float4* s = (const float4*)src + (size_t)i * 2;
    float4 a = s[0], b = s[1];
    uint4 o;
    o.x = (unsigned)f2bf(a.x) | ((unsigned)f2bf(a.y) << 16);
    o.y = (unsigned)f2bf(a.z) | ((unsigned)f2bf(a.w) << 16);
    o.z = (unsigned)f2bf(b.x) | ((unsigned)f2bf(b.y) << 16);
    o.w = (unsigned)f2bf(b.z) | ((unsigned)f2bf(b.w) << 16);
    ((uint4*)dst)[i] = o;
}

// ---------------------------------------------------------------------------
// Weight convert + transpose:  dst[n*K + k] (bf16)
//   tde==0: src row-major [K,N]                -> src[k*N + n]
//   tde==1: src = W[t,d,e], n = t*128+e, k = d -> src[t*16384 + k*128 + e]
// ---------------------------------------------------------------------------
__global__ void convw_kernel(const float* __restrict__ src, u16* __restrict__ dst,
                             int K, int N, int tde)
{
    int i = blockIdx.x * blockDim.x + threadIdx.x;
    if (i >= K * N) return;
    int n = i / K, k = i - n * K;
    size_t si;
    if (tde) {
        int t = n >> 7, e = n & 127;
        si = (size_t)t * 16384 + (size_t)k * 128 + e;
    } else {
        si = (size_t)k * N + n;
    }
    dst[i] = f2bf(src[si]);
}

// ---------------------------------------------------------------------------
// Tiled WMMA GEMM on bf16:  C = act(A[M,K] @ Bt[N,K]^T + bias)
// Block tile 128x128x32, 8 waves (4 M x 2 N), each wave 2x4 tiles of 16x16.
// Staging: global_load_async_to_lds_b128 (2x16B chunks/thread per matrix/step).
//   OUT_BF16 -> Cb (bf16), OUT_F32 -> Cf (fp32), may be combined.
//   BIL      -> fused biaffine: bil[m, n>>7] += C[m,n]*lh[m, n&127] (atomic)
// ---------------------------------------------------------------------------
#define BM 128
#define BN 128
#define BK 32
#define LPITCH 40  // shorts: 80B pitch, 16B-aligned chunks

template<bool RELU, bool BIAS, bool OUT_BF16, bool OUT_F32, bool BIL>
__global__ __launch_bounds__(256)
void gemm_wmma_kernel(const u16* __restrict__ A,
                      const u16* __restrict__ Bt,
                      const float* __restrict__ bias,
                      const float* __restrict__ lh,
                      u16* __restrict__ Cb,
                      float* __restrict__ Cf,
                      int M, int N, int K, int T)
{
    __shared__ u16 As[BM][LPITCH];
    __shared__ u16 Bs[BN][LPITCH];

    const int tid  = threadIdx.x;
    const int lane = tid & 31;
    const int wave = tid >> 5;
    const int wm   = wave & 3;   // 0..3 : M
    const int wn   = wave >> 2;  // 0..1 : N
    const int half = lane >> 4;
    const int lm   = lane & 15;

    const int n0 = blockIdx.x * BN;
    const int m0 = blockIdx.y * BM;

    v8f zero = {0.f, 0.f, 0.f, 0.f, 0.f, 0.f, 0.f, 0.f};
    v8f acc[2][4];
    #pragma unroll
    for (int i = 0; i < 2; ++i)
        #pragma unroll
        for (int j = 0; j < 4; ++j) acc[i][j] = zero;

    // per-thread staging chunk: 512 chunks of 16B per 128x32 bf16 tile
    const int q  = tid;          // chunks 0..255 and 256..511
    for (int k0 = 0; k0 < K; k0 += BK) {
        #pragma unroll
        for (int i = 0; i < 2; ++i) {
            int cq = q + i * 256;              // 0..511
            int r  = cq >> 2;                  // row 0..127
            int c8 = (cq & 3) * 8;             // bf16 col {0,8,16,24}
            const u16* gA = &A [(size_t)(m0 + r) * K + k0 + c8];
            const u16* gB = &Bt[(size_t)(n0 + r) * K + k0 + c8];
            unsigned la = (unsigned)(size_t)&As[r][c8];
            unsigned lb = (unsigned)(size_t)&Bs[r][c8];
            asm volatile("global_load_async_to_lds_b128 %0, %1, off"
                         :: "v"(la), "v"((unsigned long long)(size_t)gA) : "memory");
            asm volatile("global_load_async_to_lds_b128 %0, %1, off"
                         :: "v"(lb), "v"((unsigned long long)(size_t)gB) : "memory");
        }
        if (k0 + BK < K)
            __builtin_prefetch(&A[(size_t)(m0 + (tid >> 1)) * K + k0 + BK], 0, 1);
        asm volatile("s_wait_asynccnt 0x0" ::: "memory");
        __syncthreads();

        // fragments from LDS (aligned ds_load_b128)
        FragBf fa[2], fb[4];
        #pragma unroll
        for (int s = 0; s < 2; ++s) {
            int row = wm * 32 + s * 16 + lm;
            // A 16x32 bf16: elems 0..7 -> K = half*8+i ; 8..15 -> K = 16+half*8+i
            fa[s].q[0] = *(const uint4*)&As[row][half * 8];
            fa[s].q[1] = *(const uint4*)&As[row][16 + half * 8];
        }
        #pragma unroll
        for (int s = 0; s < 4; ++s) {
            int col = wn * 64 + s * 16 + lm;
            // B 32x16 bf16: elems 0..7 -> K = half*8+i ; 8..15 -> K = 16+half*8+i
            fb[s].q[0] = *(const uint4*)&Bs[col][half * 8];
            fb[s].q[1] = *(const uint4*)&Bs[col][16 + half * 8];
        }

        #pragma unroll
        for (int sm = 0; sm < 2; ++sm)
            #pragma unroll
            for (int sn = 0; sn < 4; ++sn)
                acc[sm][sn] = __builtin_amdgcn_wmma_f32_16x16x32_bf16(
                    false, fa[sm].v, false, fb[sn].v,
                    (short)0, acc[sm][sn], false, false);

        __syncthreads();
    }

    if (!BIL) {
        #pragma unroll
        for (int sm = 0; sm < 2; ++sm) {
            #pragma unroll
            for (int sn = 0; sn < 4; ++sn) {
                int gn = n0 + wn * 64 + sn * 16 + lm;
                float bv = BIAS ? bias[gn] : 0.f;
                #pragma unroll
                for (int r = 0; r < 8; ++r) {
                    int gm = m0 + wm * 32 + sm * 16 + r + 8 * half;
                    float v = acc[sm][sn][r] + bv;
                    if (RELU) v = fmaxf(v, 0.f);
                    if (OUT_F32)  Cf[(size_t)gm * N + gn] = v;
                    if (OUT_BF16) Cb[(size_t)gm * N + gn] = f2bf(v);
                }
            }
        }
    } else {
        #pragma unroll
        for (int sm = 0; sm < 2; ++sm) {
            #pragma unroll
            for (int sn = 0; sn < 4; ++sn) {
                int gn = n0 + wn * 64 + sn * 16 + lm;
                int t  = gn >> 7;        // constant across the 16 lanes
                int e  = gn & 127;
                #pragma unroll
                for (int r = 0; r < 8; ++r) {
                    int gm = m0 + wm * 32 + sm * 16 + r + 8 * half;
                    float v = acc[sm][sn][r] * lh[(size_t)gm * 128 + e];
                    v += __shfl_xor(v, 1, 32);
                    v += __shfl_xor(v, 2, 32);
                    v += __shfl_xor(v, 4, 32);
                    v += __shfl_xor(v, 8, 32);
                    if (lm == 0)
                        atomicAdd(&Cf[(size_t)gm * T + t], v);
                }
            }
        }
    }
}

// ---------------------------------------------------------------------------
// ht[b,t] = bias[t] + sum_e lh[b,e]*Wh[e,t] + ld[b,e]*Wd[e,t]
// ---------------------------------------------------------------------------
__global__ __launch_bounds__(128)
void ht_kernel(const float* __restrict__ lh, const float* __restrict__ ld,
               const float* __restrict__ Wh, const float* __restrict__ Wd,
               const float* __restrict__ bias, float* __restrict__ ht)
{
    __shared__ float lhd[128], ldd[128];
    int b = blockIdx.x;
    int t = threadIdx.x;
    lhd[t] = lh[(size_t)b * 128 + t];
    ldd[t] = ld[(size_t)b * 128 + t];
    __syncthreads();
    if (t < 50) {
        float s = bias[t];
        #pragma unroll 4
        for (int e = 0; e < 128; ++e)
            s += lhd[e] * Wh[e * 50 + t] + ldd[e] * Wd[e * 50 + t];
        ht[(size_t)b * 50 + t] = s;
    }
}

__global__ void fill_zero_kernel(float* __restrict__ p, int n)
{
    int i = blockIdx.x * blockDim.x + threadIdx.x;
    if (i < n) p[i] = 0.f;
}

// scores[b,t,0,tt] = bil[b,t] + ht[b,tt]; two elements per thread (50 is even)
__global__ void combine_kernel(const float* __restrict__ bil,
                               const float* __restrict__ ht,
                               float2* __restrict__ out, int total2)
{
    int i = blockIdx.x * blockDim.x + threadIdx.x;
    if (i < total2) {
        int idx = i * 2;
        int b   = idx / 2500;
        int rem = idx - b * 2500;
        int t   = rem / 50;
        int tt  = rem - t * 50;
        float base = bil[b * 50 + t];
        out[i] = make_float2(base + ht[b * 50 + tt], base + ht[b * 50 + tt + 1]);
    }
}

// ---------------------------------------------------------------------------
extern "C" void kernel_launch(void* const* d_in, const int* in_sizes, int n_in,
                              void* d_out, int out_size, void* d_ws, size_t ws_size,
                              hipStream_t stream) {
    (void)in_sizes; (void)n_in; (void)out_size; (void)ws_size;

    const int B = 32768, T = 50;

    const float* heads = (const float*)d_in[0];
    const float* deps  = (const float*)d_in[1];
    const float* hW1 = (const float*)d_in[2];  const float* hb1 = (const float*)d_in[3];
    const float* hW2 = (const float*)d_in[4];  const float* hb2 = (const float*)d_in[5];
    const float* hW3 = (const float*)d_in[6];  const float* hb3 = (const float*)d_in[7];
    const float* dW1 = (const float*)d_in[8];  const float* db1 = (const float*)d_in[9];
    const float* dW2 = (const float*)d_in[10]; const float* db2 = (const float*)d_in[11];
    const float* dW3 = (const float*)d_in[12]; const float* db3 = (const float*)d_in[13];
    const float* W   = (const float*)d_in[14];
    const float* Wh  = (const float*)d_in[15];
    const float* Wd  = (const float*)d_in[16];
    const float* bv  = (const float*)d_in[17];

    // ---- workspace carve-out (256B aligned) ----
    char* cur = (char*)d_ws;
    auto alloc = [&](size_t bytes) {
        char* p = cur;
        cur += (bytes + 255) & ~(size_t)255;
        return (void*)p;
    };
    u16*   headsb = (u16*)alloc((size_t)B * 768 * 2);
    u16*   depsb  = (u16*)alloc((size_t)B * 768 * 2);
    u16*   buf1b  = (u16*)alloc((size_t)B * 512 * 2);   // reused head/dep
    u16*   buf2b  = (u16*)alloc((size_t)B * 256 * 2);
    float* lhf    = (float*)alloc((size_t)B * 128 * 4);
    float* ldf    = (float*)alloc((size_t)B * 128 * 4);
    u16*   ldb    = (u16*)alloc((size_t)B * 128 * 2);
    float* bil    = (float*)alloc((size_t)B * T * 4);
    float* ht     = (float*)alloc((size_t)B * T * 4);
    u16* hW1t = (u16*)alloc(768 * 512 * 2);
    u16* hW2t = (u16*)alloc(512 * 256 * 2);
    u16* hW3t = (u16*)alloc(256 * 128 * 2);
    u16* dW1t = (u16*)alloc(768 * 512 * 2);
    u16* dW2t = (u16*)alloc(512 * 256 * 2);
    u16* dW3t = (u16*)alloc(256 * 128 * 2);
    u16* Wt   = (u16*)alloc((size_t)T * 128 * 128 * 2);

    dim3 blk(256);

    // ---- conversions ----
    {
        int n8 = B * 768 / 8;
        convv_kernel<<<(n8 + 255) / 256, blk, 0, stream>>>(heads, headsb, n8);
        convv_kernel<<<(n8 + 255) / 256, blk, 0, stream>>>(deps,  depsb,  n8);
    }
    auto convw = [&](const float* s, u16* d, int K, int N, int tde) {
        int n = K * N;
        convw_kernel<<<(n + 255) / 256, blk, 0, stream>>>(s, d, K, N, tde);
    };
    convw(hW1, hW1t, 768, 512, 0);
    convw(hW2, hW2t, 512, 256, 0);
    convw(hW3, hW3t, 256, 128, 0);
    convw(dW1, dW1t, 768, 512, 0);
    convw(dW2, dW2t, 512, 256, 0);
    convw(dW3, dW3t, 256, 128, 0);
    convw(W,   Wt,   128, T * 128, 1);

    // ---- head MLP ----
    gemm_wmma_kernel<true,  true,  true,  false, false><<<dim3(512 / BN, B / BM), blk, 0, stream>>>(
        headsb, hW1t, hb1, nullptr, buf1b, nullptr, B, 512, 768, T);
    gemm_wmma_kernel<true,  true,  true,  false, false><<<dim3(256 / BN, B / BM), blk, 0, stream>>>(
        buf1b, hW2t, hb2, nullptr, buf2b, nullptr, B, 256, 512, T);
    gemm_wmma_kernel<false, true,  false, true,  false><<<dim3(128 / BN, B / BM), blk, 0, stream>>>(
        buf2b, hW3t, hb3, nullptr, nullptr, lhf, B, 128, 256, T);

    // ---- dep MLP (reuses buf1b/buf2b); layer 3 dual-writes fp32 + bf16 ----
    gemm_wmma_kernel<true,  true,  true,  false, false><<<dim3(512 / BN, B / BM), blk, 0, stream>>>(
        depsb, dW1t, db1, nullptr, buf1b, nullptr, B, 512, 768, T);
    gemm_wmma_kernel<true,  true,  true,  false, false><<<dim3(256 / BN, B / BM), blk, 0, stream>>>(
        buf1b, dW2t, db2, nullptr, buf2b, nullptr, B, 256, 512, T);
    gemm_wmma_kernel<false, true,  true,  true,  false><<<dim3(128 / BN, B / BM), blk, 0, stream>>>(
        buf2b, dW3t, db3, nullptr, ldb, ldf, B, 128, 256, T);

    // ---- fused biaffine: bil[b,t] = sum_e (ld @ W[t])[b,e] * lh[b,e] ----
    int nbil = B * T;
    fill_zero_kernel<<<(nbil + 255) / 256, blk, 0, stream>>>(bil, nbil);
    gemm_wmma_kernel<false, false, false, false, true><<<dim3((T * 128) / BN, B / BM), blk, 0, stream>>>(
        ldb, Wt, nullptr, lhf, nullptr, bil, B, T * 128, 128, T);

    // ---- h_term + d_term + bias ----
    ht_kernel<<<B, 128, 0, stream>>>(lhf, ldf, Wh, Wd, bv, ht);

    // ---- broadcast add -> [B, T, 1, T] ----
    int total2 = B * T * T / 2;
    combine_kernel<<<(total2 + 255) / 256, blk, 0, stream>>>(bil, ht, (float2*)d_out, total2);
}
